// TreeLSTM_45174466020016
// MI455X (gfx1250) — matrix operand order repaired and checked by reading, not compiled, not run on hardware
//
#include <hip/hip_runtime.h>
#include <hip/hip_bf16.h>

typedef __bf16 bf16_t;
typedef __attribute__((ext_vector_type(16))) __bf16 v16bf;
typedef __attribute__((ext_vector_type(8)))  float  v8f;

union Frag16 {
    v16bf v;
    bf16_t e[16];
    int4  q[2];
};

__device__ __forceinline__ float sigmoidf_(float x) {
    return 1.0f / (1.0f + __expf(-x));
}
__device__ __forceinline__ float tanhf_(float x) {
    // tanh(x) = 1 - 2/(exp(2x)+1)  -> stays on the VALU exp path
    return 1.0f - 2.0f / (__expf(2.0f * x) + 1.0f);
}

// ---------------------------------------------------------------------------
// Pack a row-major f32 weight matrix W[Ksrc x Ncols] into the wave32 WMMA
// B-fragment layout for v_wmma_*_16x16x32_bf16, zero-padding K up to Kpad.
// Fragment (kt, nt): 32 lanes x 16 bf16, lane L covers N = nt*16 + (L&15),
// K = kt*32 + (e>=8 ? 16 : 0) + (L>=16 ? 8 : 0) + (e&7)  (e = 0..15, element order == VGPR order).
// dst index = (((nt*KT + kt)*32 + lane)*16 + e), KT = Kpad/32.
// ---------------------------------------------------------------------------
__global__ void pack_b_kernel(const float* __restrict__ W, bf16_t* __restrict__ dst,
                              int Ksrc, int Kpad, int Ncols)
{
    int idx = blockIdx.x * blockDim.x + threadIdx.x;
    int total = Kpad * Ncols;
    if (idx >= total) return;
    int e    = idx & 15;
    int lane = (idx >> 4) & 31;
    int t    = idx >> 9;            // nt*KT + kt
    int KT   = Kpad >> 5;
    int kt   = t % KT;
    int nt   = t / KT;
    int K    = kt * 32 + ((e >> 3) << 4) + ((lane >> 4) << 3) + (e & 7);
    int n    = nt * 16 + (lane & 15);
    float v  = (K < Ksrc) ? W[(size_t)K * Ncols + n] : 0.0f;
    dst[idx] = (bf16_t)v;
}

// ---------------------------------------------------------------------------
// Leaf: hc = x @ W_leaf + b_leaf ;  h = hc[:, :128] (bf16), c = hc[:, 128:] (f32)
// x: [32768 x 300] f32, K padded to 320.
// Block = 256 threads = 8 waves = one 16-row M-tile; wave w computes the two
// tiles {nt=w, nt=w+8}: one A fragment feeds two independent accumulators,
// tile w is exactly the h columns / tile w+8 exactly the c columns.
// A-fragment for kt+1 is loaded/converted BEFORE the WMMAs consuming A[kt]
// (software pipeline: shrinks WMMA->VALU WAR hazard windows).
// ---------------------------------------------------------------------------
__device__ __forceinline__ void load_a_leaf(const float* __restrict__ xr,
                                            int kt, int hi, Frag16& a)
{
    const int D = 300;
    int base = kt * 32 + hi * 8;
#pragma unroll
    for (int e = 0; e < 8; e++) {
        int k = base + e;                 // guards fold at compile time
        a.e[e] = (k < D) ? (bf16_t)xr[k] : (bf16_t)0.0f;
    }
#pragma unroll
    for (int e = 0; e < 8; e++) {
        int k = base + 16 + e;
        a.e[8 + e] = (k < D) ? (bf16_t)xr[k] : (bf16_t)0.0f;
    }
}

__global__ __launch_bounds__(256) void leaf_kernel(
    const float* __restrict__ x, const bf16_t* __restrict__ wpk,
    const float* __restrict__ b_leaf,
    bf16_t* __restrict__ h_out, float* __restrict__ c_out)
{
    const int KT = 10, H = 128;
    int wave = threadIdx.x >> 5;
    int lane = threadIdx.x & 31;
    int hi   = lane >> 4;
    int ln   = lane & 15;
    int mt   = blockIdx.x;
    int nt0  = wave;                         // h-columns tile
    int nt1  = wave + 8;                     // c-columns tile
    int row  = mt * 16 + ln;                 // A-fragment row for this lane
    const float* xr = x + (size_t)row * 300;

    v8f acc0, acc1;
    {
        float b0 = b_leaf[nt0 * 16 + ln];
        float b1 = b_leaf[nt1 * 16 + ln];
#pragma unroll
        for (int r = 0; r < 8; r++) { acc0[r] = b0; acc1[r] = b1; }
    }

    Frag16 a_cur, a_nxt;
    load_a_leaf(xr, 0, hi, a_cur);
#pragma unroll
    for (int kt = 0; kt < KT; kt++) {
        Frag16 b0, b1;
        const int4* bp0 = (const int4*)(wpk + (((size_t)nt0 * KT + kt) * 32 + lane) * 16);
        const int4* bp1 = (const int4*)(wpk + (((size_t)nt1 * KT + kt) * 32 + lane) * 16);
        b0.q[0] = bp0[0]; b0.q[1] = bp0[1];
        b1.q[0] = bp1[0]; b1.q[1] = bp1[1];
        if (kt + 1 < KT) load_a_leaf(xr, kt + 1, hi, a_nxt);   // prefetch next A
        acc0 = __builtin_amdgcn_wmma_f32_16x16x32_bf16(
            false, a_cur.v, false, b0.v, (short)0, acc0, false, false);
        acc1 = __builtin_amdgcn_wmma_f32_16x16x32_bf16(
            false, a_cur.v, false, b1.v, (short)0, acc1, false, false);
        if (kt + 1 < KT) a_cur = a_nxt;
    }

    int col = nt0 * 16 + ln;                 // in [0,128): h col == c col
#pragma unroll
    for (int r = 0; r < 8; r++) {
        int ro = mt * 16 + r + hi * 8;       // C/D layout: vgpr r -> row r (+8 hi half)
        h_out[(size_t)ro * H + col] = (bf16_t)acc0[r];
        c_out[(size_t)ro * H + col] = acc1[r];
    }
}

// ---------------------------------------------------------------------------
// Per-wave TreeLSTM reduce for one 16-row M-tile x one 16-wide h-column block
// (colt). Gate tiles {colt, colt+8, .., colt+32} == all 5 gates for those
// columns -> 5 register accumulators, elementwise entirely in registers.
// Two passes of 5 independent WMMAs (left child then right child) keep
// same-accumulator WMMAs >= 5 instructions apart (hazard distance covered by
// useful work). h pointers are generic: callers pass global OR LDS memory.
// Partial tiles: A addresses clamped (EXEC all-1s for WMMA), stores guarded.
// ---------------------------------------------------------------------------
__device__ __forceinline__ void level_reduce_wave(
    const bf16_t* h_in, const float* c_in,
    const bf16_t* __restrict__ wl, const bf16_t* __restrict__ wr,
    const float* __restrict__ bias,
    bf16_t* h_out, float* c_out,
    int m0, int lane, int colt, int M, int Nin, int nsh)
{
    const int H = 128, KT = 4;
    int hi = lane >> 4;
    int ln = lane & 15;

    int m = m0 + ln;                         // A-fragment row (clamped)
    if (m >= M) m = M - 1;
    int bidx = m >> nsh;
    int nidx = m & ((1 << nsh) - 1);
    size_t lrow = (size_t)bidx * Nin + 2 * nidx;
    const bf16_t* pl = h_in + lrow * H;
    const bf16_t* pr = h_in + (lrow + 1) * H;

    v8f acc[5];
#pragma unroll
    for (int g = 0; g < 5; g++) {
        float bv = bias[g * H + colt * 16 + ln];
#pragma unroll
        for (int r = 0; r < 8; r++) acc[g][r] = bv;
    }

#pragma unroll
    for (int kt = 0; kt < KT; kt++) {
        int base = kt * 32 + hi * 8;
        Frag16 al, ar;
        al.q[0] = *(const int4*)(pl + base);
        al.q[1] = *(const int4*)(pl + base + 16);
        ar.q[0] = *(const int4*)(pr + base);
        ar.q[1] = *(const int4*)(pr + base + 16);
        // Pass 1: left child, 5 independent WMMAs
#pragma unroll
        for (int g = 0; g < 5; g++) {
            int nt = g * 8 + colt;
            Frag16 bl;
            const int4* blp = (const int4*)(wl + (((size_t)nt * KT + kt) * 32 + lane) * 16);
            bl.q[0] = blp[0]; bl.q[1] = blp[1];
            acc[g] = __builtin_amdgcn_wmma_f32_16x16x32_bf16(
                false, al.v, false, bl.v, (short)0, acc[g], false, false);
        }
        // Pass 2: right child, 5 independent WMMAs
#pragma unroll
        for (int g = 0; g < 5; g++) {
            int nt = g * 8 + colt;
            Frag16 br;
            const int4* brp = (const int4*)(wr + (((size_t)nt * KT + kt) * 32 + lane) * 16);
            br.q[0] = brp[0]; br.q[1] = brp[1];
            acc[g] = __builtin_amdgcn_wmma_f32_16x16x32_bf16(
                false, ar.v, false, br.v, (short)0, acc[g], false, false);
        }
    }

    // Elementwise LSTM cell update, entirely in registers.
    int col = colt * 16 + ln;
#pragma unroll
    for (int r = 0; r < 8; r++) {
        int me = m0 + r + hi * 8;
        if (me >= M) continue;
        int be = me >> nsh;
        int ne = me & ((1 << nsh) - 1);
        size_t lre = (size_t)be * Nin + 2 * ne;
        float cl = c_in[lre * H + col];
        float cr = c_in[(lre + 1) * H + col];
        float iv  = sigmoidf_(acc[0][r]);
        float flv = sigmoidf_(acc[1][r]);
        float frv = sigmoidf_(acc[2][r]);
        float ov  = sigmoidf_(acc[3][r]);
        float gv  = tanhf_(acc[4][r]);
        float cn  = flv * cl + frv * cr + iv * gv;
        float hn  = ov * tanhf_(cn);
        c_out[(size_t)me * H + col] = cn;
        h_out[(size_t)me * H + col] = (bf16_t)hn;
    }
}

// One tree level in global memory (levels 0..7: M = 16384 .. 128 rows).
// Block = one 16-row M-tile, 8 waves; wave w owns h-columns [w*16, w*16+16).
__global__ __launch_bounds__(256) void reduce_kernel(
    const bf16_t* __restrict__ h_in, const float* __restrict__ c_in,
    const bf16_t* __restrict__ wl, const bf16_t* __restrict__ wr,
    const float* __restrict__ bias,
    bf16_t* __restrict__ h_out, float* __restrict__ c_out,
    int M, int Nin, int nsh /* log2(Nout) */)
{
    int wave = threadIdx.x >> 5;
    int lane = threadIdx.x & 31;
    level_reduce_wave(h_in, c_in, wl, wr, bias, h_out, c_out,
                      blockIdx.x * 16, lane, wave, M, Nin, nsh);
}

// ---------------------------------------------------------------------------
// Fused top of the tree: levels with Nin = 16, 8, 4, 2 (M = 64, 32, 16, 8)
// in ONE persistent 1024-thread workgroup -> kills 3 extra kernel dispatches
// at the latency-bound tip of the tree.
// h ping-pongs in LDS (A-fragments become ds_load_b128); c ping-pongs in
// L2-resident global scratch with __threadfence + __syncthreads between
// levels. Wave (mtile, colt) = (w>>3, w&7); waves beyond the level's m-tile
// count idle through the barriers.
// ---------------------------------------------------------------------------
__global__ __launch_bounds__(1024) void top_kernel(
    const bf16_t* __restrict__ h_in_g, const float* __restrict__ c0,
    float* __restrict__ c1,
    const bf16_t* __restrict__ wl, const bf16_t* __restrict__ wr,
    const float* __restrict__ bias, float* __restrict__ out)
{
    __shared__ __align__(16) bf16_t hs0[8 * 16 * 128];   // 32 KB: level-8 input
    __shared__ __align__(16) bf16_t hs1[8 * 8 * 128];    // 16 KB: ping-pong

    int tid  = threadIdx.x;
    int lane = tid & 31;
    int w    = tid >> 5;                     // wave 0..31
    int mtile = w >> 3;
    int colt  = w & 7;

    // Stage level-8 h input (8*16*128 bf16 = 2048 int4) into LDS.
    {
        const int4* src = (const int4*)h_in_g;
        int4* dst = (int4*)hs0;
        for (int i = tid; i < 2048; i += 1024) dst[i] = src[i];
    }
    __syncthreads();

    bf16_t* hin = hs0;  bf16_t* hot = hs1;
    const float* cin = c0;
    float* cot = c1;
    int Nin = 16, nsh = 3;
#pragma unroll
    for (int lev = 0; lev < 4; lev++) {
        int Nout = Nin >> 1;
        int M = 8 * Nout;
        int m_tiles = (M + 15) >> 4;
        if (mtile < m_tiles)
            level_reduce_wave(hin, cin, wl, wr, bias, hot, cot,
                              mtile * 16, lane, colt, M, Nin, nsh);
        __threadfence();                     // make global c stores visible
        __syncthreads();                     // and order LDS h across waves
        bf16_t* th = hin; hin = hot; hot = th;
        const float* tc = cin; cin = cot; cot = (float*)tc;
        Nin = Nout; nsh--;
    }

    // Root hidden state: rows 0..7 of hin -> f32 output [8,128].
    if (tid < 1024) out[tid] = (float)hin[tid];
}

extern "C" void kernel_launch(void* const* d_in, const int* in_sizes, int n_in,
                              void* d_out, int out_size, void* d_ws, size_t ws_size,
                              hipStream_t stream) {
    const float* x      = (const float*)d_in[0];   // [8, 4096, 300]
    const float* W_leaf = (const float*)d_in[1];   // [300, 256]
    const float* b_leaf = (const float*)d_in[2];   // [256]
    const float* W_l    = (const float*)d_in[3];   // [128, 640]
    const float* W_r    = (const float*)d_in[4];   // [128, 640]
    const float* bgate  = (const float*)d_in[5];   // [640]

    const int B = 8, L = 4096, H = 128;

    // Workspace carve-up
    char* ws = (char*)d_ws;
    size_t off = 0;
    auto carve = [&](size_t bytes) -> void* {
        void* p = ws + off;
        off += (bytes + 255) & ~(size_t)255;
        return p;
    };
    bf16_t* wleaf_pk = (bf16_t*)carve((size_t)320 * 256 * sizeof(bf16_t));
    bf16_t* wl_pk    = (bf16_t*)carve((size_t)128 * 640 * sizeof(bf16_t));
    bf16_t* wr_pk    = (bf16_t*)carve((size_t)128 * 640 * sizeof(bf16_t));
    bf16_t* hA       = (bf16_t*)carve((size_t)B * L * H * sizeof(bf16_t));        // 8 MB
    bf16_t* hB       = (bf16_t*)carve((size_t)B * (L / 2) * H * sizeof(bf16_t));  // 4 MB
    float*  cA       = (float*) carve((size_t)B * L * H * sizeof(float));         // 16 MB
    float*  cB       = (float*) carve((size_t)B * (L / 2) * H * sizeof(float));   // 8 MB

    // 1) Pack weights into WMMA B-fragment layout (bf16)
    {
        int tot = 320 * 256;
        pack_b_kernel<<<(tot + 255) / 256, 256, 0, stream>>>(W_leaf, wleaf_pk, 300, 320, 256);
        tot = 128 * 640;
        pack_b_kernel<<<(tot + 255) / 256, 256, 0, stream>>>(W_l, wl_pk, 128, 128, 640);
        pack_b_kernel<<<(tot + 255) / 256, 256, 0, stream>>>(W_r, wr_pk, 128, 128, 640);
    }

    // 2) Leaf transform: (B*L) x 300 @ 300 x 256 -> h (bf16), c (f32)
    {
        int m_tiles = (B * L) / 16;          // 2048 blocks, one m-tile each
        leaf_kernel<<<m_tiles, 256, 0, stream>>>(x, wleaf_pk, b_leaf, hA, cA);
    }

    // 3) Levels 0..7 (Nin 4096 -> 16), ping-pong buffers
    bf16_t* h_in = hA;  float* c_in = cA;
    bf16_t* h_ot = hB;  float* c_ot = cB;
    int Nin = L;
    for (int lev = 0; lev < 8; lev++) {
        int Nout = Nin >> 1;
        int M = B * Nout;
        int nsh = 0;
        while ((1 << nsh) < Nout) nsh++;
        int m_tiles = (M + 15) / 16;
        reduce_kernel<<<m_tiles, 256, 0, stream>>>(h_in, c_in, wl_pk, wr_pk, bgate,
                                                   h_ot, c_ot, M, Nin, nsh);
        bf16_t* th = h_in; h_in = h_ot; h_ot = th;
        float*  tc = c_in; c_in = c_ot; c_ot = tc;
        Nin = Nout;
    }
    // h_in/c_in now hold the Nin=16 level inputs.

    // 4) Fused top levels (Nin 16 -> 1) + root output, one workgroup.
    top_kernel<<<1, 1024, 0, stream>>>(h_in, c_in, c_ot, wl_pk, wr_pk, bgate,
                                       (float*)d_out);
}
